// SelfAttention1d_72327249265251
// MI455X (gfx1250) — matrix operand order repaired and checked by reading, not compiled, not run on hardware
//
#include <hip/hip_runtime.h>

#define BB 16
#define CC 512
#define LL 1024

typedef _Float16 v16h __attribute__((ext_vector_type(16)));
typedef _Float16 v8h  __attribute__((ext_vector_type(8)));
typedef float    v8f  __attribute__((ext_vector_type(8)));

// ---------------------------------------------------------------------------
// BatchNorm statistics: one block per channel, reduce over B*L = 16384 elems.
// Folds gamma/beta: hn = x*scale[c] + shift[c]
// ---------------------------------------------------------------------------
__global__ __launch_bounds__(256)
void bn_stats_kernel(const float* __restrict__ x,
                     const float* __restrict__ gamma,
                     const float* __restrict__ beta,
                     float* __restrict__ scale,
                     float* __restrict__ shift)
{
    const int c   = blockIdx.x;
    const int tid = threadIdx.x;
    float s = 0.f, s2 = 0.f;
    const float* xc = x + (size_t)c * LL;
    for (int i = tid; i < BB * LL; i += 256) {
        int b = i >> 10, l = i & (LL - 1);
        float v = xc[(size_t)b * CC * LL + l];
        s += v; s2 += v * v;
    }
    __shared__ float r1[256], r2[256];
    r1[tid] = s; r2[tid] = s2;
    __syncthreads();
    for (int st = 128; st > 0; st >>= 1) {
        if (tid < st) { r1[tid] += r1[tid + st]; r2[tid] += r2[tid + st]; }
        __syncthreads();
    }
    if (tid == 0) {
        const float inv = 1.f / (float)(BB * LL);
        float mean = r1[0] * inv;
        float var  = r2[0] * inv - mean * mean;
        float rstd = rsqrtf(var + 1e-5f);
        float sc = gamma[c] * rstd;
        scale[c] = sc;
        shift[c] = beta[c] - mean * sc;
    }
}

// ---------------------------------------------------------------------------
// Convert the four 512x512 weight matrices f32 -> f16 (row-major kept).
// ---------------------------------------------------------------------------
__global__ __launch_bounds__(256)
void cvt_weights_kernel(const float* __restrict__ w0, const float* __restrict__ w1,
                        const float* __restrict__ w2, const float* __restrict__ w3,
                        _Float16* __restrict__ dst)
{
    int i = blockIdx.x * 256 + threadIdx.x;     // 0 .. 4*262144-1
    int sel = i >> 18;
    int off = i & 0x3FFFF;
    const float* src = (sel == 0) ? w0 : (sel == 1) ? w1 : (sel == 2) ? w2 : w3;
    dst[i] = (_Float16)src[off];
}

// ---------------------------------------------------------------------------
// Normalize + transpose: x[b][c][l] f32 -> hnT[b][l][c] f16 via LDS tile.
// grid (C/32, L/32, B), block (32, 8).
// ---------------------------------------------------------------------------
__global__ __launch_bounds__(256)
void norm_transpose_kernel(const float* __restrict__ x,
                           const float* __restrict__ scale,
                           const float* __restrict__ shift,
                           _Float16* __restrict__ hnT)
{
    __shared__ _Float16 tile[32][33];
    const int b  = blockIdx.z;
    const int c0 = blockIdx.x * 32;
    const int l0 = blockIdx.y * 32;
#pragma unroll
    for (int r = 0; r < 4; r++) {
        int c = c0 + threadIdx.y + 8 * r;
        int l = l0 + threadIdx.x;
        float v = x[((size_t)b * CC + c) * LL + l];
        tile[threadIdx.y + 8 * r][threadIdx.x] = (_Float16)(v * scale[c] + shift[c]);
    }
    __syncthreads();
#pragma unroll
    for (int r = 0; r < 4; r++) {
        int l = l0 + threadIdx.y + 8 * r;
        int c = c0 + threadIdx.x;
        hnT[((size_t)b * LL + l) * CC + c] = tile[threadIdx.x][threadIdx.y + 8 * r];
    }
}

// ---------------------------------------------------------------------------
// Row softmax: S f32 [B*L rows of 1024] -> P f16. One block (256 thr) per row.
// ---------------------------------------------------------------------------
__global__ __launch_bounds__(256)
void softmax_rows_kernel(const float* __restrict__ S, _Float16* __restrict__ P)
{
    const size_t row = blockIdx.x;
    const float* s = S + row * LL;
    _Float16*    p = P + row * LL;
    const int tid = threadIdx.x;

    float4 v = reinterpret_cast<const float4*>(s)[tid];
    float m = fmaxf(fmaxf(v.x, v.y), fmaxf(v.z, v.w));

    __shared__ float red[256];
    red[tid] = m; __syncthreads();
    for (int st = 128; st > 0; st >>= 1) {
        if (tid < st) red[tid] = fmaxf(red[tid], red[tid + st]);
        __syncthreads();
    }
    m = red[0]; __syncthreads();

    float e0 = __expf(v.x - m), e1 = __expf(v.y - m);
    float e2 = __expf(v.z - m), e3 = __expf(v.w - m);
    red[tid] = e0 + e1 + e2 + e3; __syncthreads();
    for (int st = 128; st > 0; st >>= 1) {
        if (tid < st) red[tid] += red[tid + st];
        __syncthreads();
    }
    float inv = 1.0f / red[0];
    p[4 * tid + 0] = (_Float16)(e0 * inv);
    p[4 * tid + 1] = (_Float16)(e1 * inv);
    p[4 * tid + 2] = (_Float16)(e2 * inv);
    p[4 * tid + 3] = (_Float16)(e3 * inv);
}

// ---------------------------------------------------------------------------
// WMMA GEMM:  D = epilogue( A[MxK] * B[KxN] )
//   A: f16 row-major (element (m,k) at A[m*lda + k])
//   B: f16 K-major   (element (k,n) at B[n*ldb + k])
// Block = 4 waves; wave tile 64x64 (4x4 WMMA tiles); block tile 128x128.
// All epilogues use the packed transposed store: lane's 8 acc values are
// contiguous in M, stored at D[n*ldd + m] (16B f16 / 32B f32, aligned).
// MODE 0: f16  D[n*ldd+m] = acc*scale (+ bias[m] if HASB)
// MODE 1: f16  D[n*ldd+m] = acc + bias[n]
// MODE 2: f32  D[n*ldd+m] = acc*scale
// MODE 3: f32  D[n*ldd+m] = X[n*ldd+m] + bias[n] + acc
// ---------------------------------------------------------------------------
template <int MODE, bool HASB>
__global__ __launch_bounds__(128)
void wmma_gemm_kernel(const _Float16* __restrict__ A, int lda, size_t sA,
                      const _Float16* __restrict__ Bm, int ldb, size_t sB,
                      void* __restrict__ Dv, int ldd, size_t sD,
                      int K, const float* __restrict__ bias, float scale,
                      const float* __restrict__ Xadd, size_t sX)
{
    const int lane = threadIdx.x & 31;
    const int wave = threadIdx.x >> 5;
    const int bz   = blockIdx.z;
    const _Float16* Ab = A  + (size_t)bz * sA;
    const _Float16* Bb = Bm + (size_t)bz * sB;
    const int m0 = blockIdx.x * 128 + (wave >> 1) * 64;
    const int n0 = blockIdx.y * 128 + (wave & 1) * 64;
    const int lr = lane & 15;       // row (A) / column (B) within tile
    const int lh = lane >> 4;       // half-wave select

    v8f acc[4][4] = {};

    for (int kk = 0; kk < K; kk += 32) {
        v16h afrag[4], bfrag[4];
        // A fragment: lane holds row m0+16mi+lr, K halves {k0..k0+7, k0+16..k0+23}
#pragma unroll
        for (int mi = 0; mi < 4; mi++) {
            const _Float16* ap = Ab + (size_t)(m0 + 16 * mi + lr) * lda + (kk + 8 * lh);
            v8h lo = *reinterpret_cast<const v8h*>(ap);
            v8h hi = *reinterpret_cast<const v8h*>(ap + 16);
#pragma unroll
            for (int i = 0; i < 8; i++) { afrag[mi][i] = lo[i]; afrag[mi][i + 8] = hi[i]; }
        }
        // B fragment: lane holds column n0+16ni+lr, K halves {k0..k0+7, k0+16..k0+23}
#pragma unroll
        for (int ni = 0; ni < 4; ni++) {
            const _Float16* bp = Bb + (size_t)(n0 + 16 * ni + lr) * ldb + (kk + 8 * lh);
            v8h lo = *reinterpret_cast<const v8h*>(bp);
            v8h hi = *reinterpret_cast<const v8h*>(bp + 16);
#pragma unroll
            for (int i = 0; i < 8; i++) { bfrag[ni][i] = lo[i]; bfrag[ni][i + 8] = hi[i]; }
        }
#pragma unroll
        for (int mi = 0; mi < 4; mi++)
#pragma unroll
            for (int ni = 0; ni < 4; ni++)
                acc[mi][ni] = __builtin_amdgcn_wmma_f32_16x16x32_f16(
                    false, afrag[mi], false, bfrag[ni],
                    (short)0, acc[mi][ni], false, false);
    }

    // Epilogue: lane covers M = Mb..Mb+7 (Mb = tile_m + 8*lh), N = tile_n + lr.
#pragma unroll
    for (int mi = 0; mi < 4; mi++)
#pragma unroll
    for (int ni = 0; ni < 4; ni++) {
        const int Mb = m0 + 16 * mi + 8 * lh;
        const int Nn = n0 + 16 * ni + lr;
        if (MODE == 0) {
            _Float16* D = (_Float16*)Dv + (size_t)bz * sD;
            v8h o;
#pragma unroll
            for (int r = 0; r < 8; r++) {
                float bv = HASB ? bias[Mb + r] : 0.f;
                o[r] = (_Float16)(acc[mi][ni][r] * scale + bv);
            }
            *reinterpret_cast<v8h*>(D + (size_t)Nn * ldd + Mb) = o;
        } else if (MODE == 1) {
            _Float16* D = (_Float16*)Dv + (size_t)bz * sD;
            const float bv = bias[Nn];
            v8h o;
#pragma unroll
            for (int r = 0; r < 8; r++)
                o[r] = (_Float16)(acc[mi][ni][r] + bv);
            *reinterpret_cast<v8h*>(D + (size_t)Nn * ldd + Mb) = o;
        } else if (MODE == 2) {
            float* D = (float*)Dv + (size_t)bz * sD;
            v8f o;
#pragma unroll
            for (int r = 0; r < 8; r++)
                o[r] = acc[mi][ni][r] * scale;
            *reinterpret_cast<v8f*>(D + (size_t)Nn * ldd + Mb) = o;
        } else {
            float* D = (float*)Dv + (size_t)bz * sD;
            const float* xb = Xadd + (size_t)bz * sX;
            const float bv = bias[Nn];
            v8f xv = *reinterpret_cast<const v8f*>(xb + (size_t)Nn * ldd + Mb);
            v8f o;
#pragma unroll
            for (int r = 0; r < 8; r++)
                o[r] = xv[r] + bv + acc[mi][ni][r];
            *reinterpret_cast<v8f*>(D + (size_t)Nn * ldd + Mb) = o;
        }
    }
}

// ---------------------------------------------------------------------------
// Host launch
// ---------------------------------------------------------------------------
extern "C" void kernel_launch(void* const* d_in, const int* in_sizes, int n_in,
                              void* d_out, int out_size, void* d_ws, size_t ws_size,
                              hipStream_t stream)
{
    (void)in_sizes; (void)n_in; (void)out_size; (void)ws_size;

    const float* x     = (const float*)d_in[0];
    const float* gamma = (const float*)d_in[1];
    const float* beta  = (const float*)d_in[2];
    const float* Wq    = (const float*)d_in[3];
    const float* bq    = (const float*)d_in[4];
    const float* Wk    = (const float*)d_in[5];
    const float* bk    = (const float*)d_in[6];
    const float* Wv    = (const float*)d_in[7];
    const float* bv    = (const float*)d_in[8];
    const float* Wp    = (const float*)d_in[9];
    const float* bp    = (const float*)d_in[10];
    float* out = (float*)d_out;

    const size_t LC   = (size_t)LL * CC;        // 524288 elements
    const size_t LLsq = (size_t)LL * LL;        // 1048576 elements

    // Workspace layout (bytes)
    char* ws = (char*)d_ws;
    const size_t off_scale = 0;                        // 512 f32
    const size_t off_shift = 2048;                     // 512 f32
    const size_t off_w     = 4096;                     // 4 x 512KB f16
    const size_t off_hnT   = off_w   + 4ull * 524288;  // 16 MB f16  (reused as hT)
    const size_t off_qT    = off_hnT + 2ull * BB * LC; // 16 MB f16 (P overlays qT+kT)
    const size_t off_kT    = off_qT  + 2ull * BB * LC; // 16 MB f16
    const size_t off_v     = off_kT  + 2ull * BB * LC; // 16 MB f16
    const size_t off_S     = off_v   + 2ull * BB * LC; // 64 MB f32

    float*     scale = (float*)(ws + off_scale);
    float*     shift = (float*)(ws + off_shift);
    _Float16*  Wq_h  = (_Float16*)(ws + off_w);
    _Float16*  Wk_h  = Wq_h + 262144;
    _Float16*  Wv_h  = Wq_h + 2 * 262144;
    _Float16*  Wp_h  = Wq_h + 3 * 262144;
    _Float16*  hnT   = (_Float16*)(ws + off_hnT);
    _Float16*  qT    = (_Float16*)(ws + off_qT);
    _Float16*  kT    = (_Float16*)(ws + off_kT);
    _Float16*  vbuf  = (_Float16*)(ws + off_v);
    float*     S     = (float*)(ws + off_S);
    _Float16*  P     = qT;    // q,k dead after S; 32MB fits qT+kT region
    _Float16*  hT    = hnT;   // hn dead after QKV

    const float attn_scale = 0.04419417382415922f;     // 512^-0.5

    // 1) BN statistics
    bn_stats_kernel<<<CC, 256, 0, stream>>>(x, gamma, beta, scale, shift);
    // 2) Weights -> f16
    cvt_weights_kernel<<<4096, 256, 0, stream>>>(Wq, Wk, Wv, Wp, Wq_h);
    // 3) Normalize + transpose -> hnT[b][l][c]
    norm_transpose_kernel<<<dim3(CC / 32, LL / 32, BB), dim3(32, 8), 0, stream>>>(
        x, scale, shift, hnT);

    // 4) Q,K projections: A=W (M=o), B=hnT (N=l) -> transposed store qT/kT[l][o]
    dim3 gQK(CC / 128, LL / 128, BB);
    wmma_gemm_kernel<0, true><<<gQK, 128, 0, stream>>>(
        Wq_h, CC, 0, hnT, CC, LC, qT, CC, LC, CC, bq, 1.f, nullptr, 0);
    wmma_gemm_kernel<0, true><<<gQK, 128, 0, stream>>>(
        Wk_h, CC, 0, hnT, CC, LC, kT, CC, LC, CC, bk, 1.f, nullptr, 0);
    // V projection: A=hnT (M=l), B=Wv^T via row-major Wv (N=o)
    //   -> transposed store lands in natural v[o][l]
    dim3 gV(LL / 128, CC / 128, BB);
    wmma_gemm_kernel<1, true><<<gV, 128, 0, stream>>>(
        hnT, CC, LC, Wv_h, CC, 0, vbuf, LL, LC, CC, bv, 1.f, nullptr, 0);

    // 5) Scores: A=kT (M=j), B=qT (N=i) -> transposed store = natural S[i][j]
    dim3 gS(LL / 128, LL / 128, BB);
    wmma_gemm_kernel<2, false><<<gS, 128, 0, stream>>>(
        kT, CC, LC, qT, CC, LC, S, LL, LLsq, CC, nullptr, attn_scale, nullptr, 0);

    // 6) Softmax over j -> P f16
    softmax_rows_kernel<<<BB * LL, 256, 0, stream>>>(S, P);

    // 7) A=v (M=c), B=P (N=i) -> transposed store hT[i][c]   (K=L)
    dim3 gPV(CC / 128, LL / 128, BB);
    wmma_gemm_kernel<0, false><<<gPV, 128, 0, stream>>>(
        vbuf, LL, LC, P, LL, LLsq, hT, CC, LC, LL, nullptr, 1.f, nullptr, 0);

    // 8) A=hT (M=i), B=Wp^T via row-major Wp (N=o) -> out[o][i] = x + bp + acc
    dim3 gOut(LL / 128, CC / 128, BB);
    wmma_gemm_kernel<3, true><<<gOut, 128, 0, stream>>>(
        hT, CC, LC, Wp_h, CC, 0, out, LL, LC, CC, bp, 1.f, x, LC);
}
